// TensorProduct_31421980738348
// MI455X (gfx1250) — matrix operand order repaired and checked by reading, not compiled, not run on hardware
//
#include <hip/hip_runtime.h>
#include <hip/hip_bf16.h>

// e3nn-style channelwise tensor product, MUL=64.
// Per-row feature vectors (scales folded) -> bf16 WMMA GEMMs against stacked
// 64x64 weights staged transposed in LDS. Memory floor ~17.7us @ 23.3 TB/s
// (2064 B/row, 200k rows); bf16 WMMA keeps the ~18 GFLOP under that floor.
// Block = 5 waves (160 thr), 153.3 KB LDS -> 2 blocks / 320 KB WGP = 10 waves.

typedef __attribute__((ext_vector_type(16))) __bf16        v16bf;
typedef __attribute__((ext_vector_type(2)))  __bf16        v2bf;
typedef __attribute__((ext_vector_type(8)))  float         v8f;
typedef __attribute__((ext_vector_type(4)))  float         v4f;
typedef __attribute__((ext_vector_type(4)))  unsigned int  v4u;

#define WAVES_PER_BLOCK 5

union Frag32 { v4u u4[2]; v16bf bf; };

__device__ __forceinline__ unsigned short f2bf(float f) {
  unsigned int u = __builtin_bit_cast(unsigned int, f);
  u = (u + 0x7FFFu + ((u >> 16) & 1u)) >> 16;   // RNE
  return (unsigned short)u;
}

__device__ __forceinline__ unsigned int pk2(float a, float b) {
#if __has_builtin(__builtin_amdgcn_cvt_pk_bf16_f32)
  v2bf t = __builtin_amdgcn_cvt_pk_bf16_f32(a, b);   // v_cvt_pk_bf16_f32 (RNE)
  return __builtin_bit_cast(unsigned int, t);
#else
  return (unsigned int)f2bf(a) | ((unsigned int)f2bf(b) << 16);
#endif
}

// LDS strides in ushort units (padded for conflict-free b128 reads)
#define WS_STRIDE 136   // 128 K + 8 pad  -> 272 B rows
#define WV_STRIDE 200   // 192 K + 8 pad  -> 400 B rows
#define FT_STRIDE 712   // 704 feats + 8  -> 1424 B rows

__global__ __launch_bounds__(32 * WAVES_PER_BLOCK)
void tp_cdna5_kernel(const float* __restrict__ x1, const float* __restrict__ x2,
                     const float* __restrict__ w_ss_s, const float* __restrict__ w_vv_s,
                     const float* __restrict__ w_sv_v, const float* __restrict__ w_vs_v,
                     const float* __restrict__ w_vv_v,
                     float* __restrict__ out, int n, int numTiles)
{
  __shared__ __attribute__((aligned(16))) unsigned short lds_ws[64 * WS_STRIDE];
  __shared__ __attribute__((aligned(16))) unsigned short lds_wv[64 * WV_STRIDE];
  __shared__ __attribute__((aligned(16))) unsigned short lds_ft[WAVES_PER_BLOCK * 16 * FT_STRIDE];

  const int tid = threadIdx.x;

  // ---- stage weights into LDS as bf16, transposed [n][k] (first 128 thr) ----
  if (tid < 128) {
    const int n0   = tid & 63;
    const int half = tid >> 6;
    const float* srcS = half ? w_vv_s : w_ss_s;
    unsigned short* wsRow = &lds_ws[n0 * WS_STRIDE + half * 64];
    #pragma unroll 8
    for (int j = 0; j < 64; ++j) wsRow[j] = f2bf(srcS[j * 64 + n0]);

    unsigned short* wvRow = &lds_wv[n0 * WV_STRIDE];
    if (half == 0) {
      #pragma unroll 8
      for (int j = 0; j < 64; ++j) wvRow[j]      = f2bf(w_vs_v[j * 64 + n0]);
      #pragma unroll 8
      for (int j = 0; j < 32; ++j) wvRow[64 + j] = f2bf(w_vv_v[j * 64 + n0]);
    } else {
      #pragma unroll 8
      for (int j = 0; j < 32; ++j) wvRow[96 + j]  = f2bf(w_vv_v[(32 + j) * 64 + n0]);
      #pragma unroll 8
      for (int j = 0; j < 64; ++j) wvRow[128 + j] = f2bf(w_sv_v[j * 64 + n0]);
    }
  }
  __syncthreads();

  const int lane = tid & 31;
  // wave id is wave-uniform: force it scalar so tile/z0/fullTile become SALU
  const int wave = __builtin_amdgcn_readfirstlane(tid >> 5);
  const int row  = lane & 15;   // M (A) and N (B/C) index
  const int h    = lane >> 4;   // half-wave split

  unsigned short* ftRow = &lds_ft[(wave * 16 + row) * FT_STRIDE];

  const float A0  = 0.08838834764831845f;       // sqrt(1/(2*64))
  const float A0V = A0 * 0.5773502691896258f;   // a0/sqrt(3)
  const float A1  = 0.07216878364870323f;       // sqrt(1/(3*64))
  const float A1C = A1 * 0.7071067811865476f;   // a1/sqrt(2)

  // A-fragment: 16x32 bf16, half-wave K layout {0-7,16-23}/{8-15,24-31}
  auto loadA = [&](int fb, int kt) -> v16bf {
    Frag32 f;
    const int o = fb + kt * 32 + h * 8;
    f.u4[0] = *(const v4u*)(ftRow + o);
    f.u4[1] = *(const v4u*)(ftRow + o + 16);
    return f.bf;
  };
  // B-fragment: 32x16 bf16, lane = N, K packed pairs, half-wave K split
  auto loadB = [&](const unsigned short* w, int stride, int kt, int nt) -> v16bf {
    Frag32 f;
    const unsigned short* p = w + (nt * 16 + row) * stride + kt * 32 + h * 8;
    f.u4[0] = *(const v4u*)(p);
    f.u4[1] = *(const v4u*)(p + 16);
    return f.bf;
  };

  for (int tile = blockIdx.x * WAVES_PER_BLOCK + wave; tile < numTiles;
       tile += gridDim.x * WAVES_PER_BLOCK) {
    const int z0 = tile * 16;                    // scalar
    const bool fullTile = (z0 + 16 <= n);        // scalar compare
    int z = z0 + row; if (z >= n) z = n - 1;     // clamp, EXEC stays all-ones

    const v4f q = __builtin_nontemporal_load((const v4f*)(x2 + (size_t)z * 4));
    const float s2 = q.x, v20 = q.y, v21 = q.z, v22 = q.w;
    const float c_ss = A0 * s2;
    const float c_vs = A1 * s2;
    const float c_sv0 = A1 * v20, c_sv1 = A1 * v21, c_sv2 = A1 * v22;

    const float* x1r = x1 + (size_t)z * 256;

    // ---- build bf16 feature rows in LDS (scales folded) ----
    #pragma unroll
    for (int c = 0; c < 8; ++c) {
      const int u0 = h * 32 + 4 * c;
      const v4f s4 = __builtin_nontemporal_load((const v4f*)(x1r + u0));
      const v4f va = __builtin_nontemporal_load((const v4f*)(x1r + 64 + 3 * u0));
      const v4f vb = __builtin_nontemporal_load((const v4f*)(x1r + 64 + 3 * u0 + 4));
      const v4f vc = __builtin_nontemporal_load((const v4f*)(x1r + 64 + 3 * u0 + 8));
      const float xs[4]  = { s4.x, s4.y, s4.z, s4.w };
      const float xv0[4] = { va.x, va.w, vb.z, vc.y };
      const float xv1[4] = { va.y, vb.x, vb.w, vc.z };
      const float xv2[4] = { va.z, vb.y, vc.x, vc.w };

      float fs1[4], fs2[4], sA[3][4], sB[3][4], sC[3][4];
      #pragma unroll
      for (int i = 0; i < 4; ++i) {
        const float d = xv0[i] * v20 + xv1[i] * v21 + xv2[i] * v22;
        fs1[i] = c_ss * xs[i];
        fs2[i] = A0V * d;
        sA[0][i] = c_vs * xv0[i];
        sA[1][i] = c_vs * xv1[i];
        sA[2][i] = c_vs * xv2[i];
        sB[0][i] = A1C * (xv1[i] * v22 - xv2[i] * v21);  // (x1v x x2v)_k
        sB[1][i] = A1C * (xv2[i] * v20 - xv0[i] * v22);
        sB[2][i] = A1C * (xv0[i] * v21 - xv1[i] * v20);
        sC[0][i] = c_sv0 * xs[i];
        sC[1][i] = c_sv1 * xs[i];
        sC[2][i] = c_sv2 * xs[i];
      }
      *(uint2*)(ftRow + u0)      = make_uint2(pk2(fs1[0], fs1[1]), pk2(fs1[2], fs1[3]));
      *(uint2*)(ftRow + 64 + u0) = make_uint2(pk2(fs2[0], fs2[1]), pk2(fs2[2], fs2[3]));
      #pragma unroll
      for (int k = 0; k < 3; ++k) {
        unsigned short* fk = ftRow + 128 + 192 * k;
        *(uint2*)(fk + u0)       = make_uint2(pk2(sA[k][0], sA[k][1]), pk2(sA[k][2], sA[k][3]));
        *(uint2*)(fk + 64 + u0)  = make_uint2(pk2(sB[k][0], sB[k][1]), pk2(sB[k][2], sB[k][3]));
        *(uint2*)(fk + 128 + u0) = make_uint2(pk2(sC[k][0], sC[k][1]), pk2(sC[k][2], sC[k][3]));
      }
    }
    // same-wave DS ops are in-order (ISA 7.3); stop compiler reordering only
    asm volatile("" ::: "memory");

    float* const outBase = out + (size_t)(z0 + h * 8) * 256;

    // ---- out_s: (16x128) x (128x64) ----
    #pragma unroll
    for (int nt = 0; nt < 4; ++nt) {
      v8f acc = {};
      #pragma unroll
      for (int kt = 0; kt < 4; ++kt) {
        v16bf a = loadA(0, kt);
        v16bf b = loadB(lds_ws, WS_STRIDE, kt, nt);
        acc = __builtin_amdgcn_wmma_f32_16x16x32_bf16(false, a, false, b,
                                                      (short)0, acc, false, false);
      }
      const int col = nt * 16 + row;
      if (fullTile) {
        float* p = outBase + col;                 // r*1024B folds into ioffset
        #pragma unroll
        for (int r = 0; r < 8; ++r) __builtin_nontemporal_store(acc[r], p + r * 256);
      } else {
        #pragma clang loop unroll(disable)
        for (int r = 0; r < 8; ++r) {
          const int zz = z0 + h * 8 + r;
          if (zz < n) out[(size_t)zz * 256 + col] = acc[r];
        }
      }
    }

    // ---- out_v[:,k]: (16x192) x (192x64), k = 0..2 ----
    #pragma unroll
    for (int k = 0; k < 3; ++k) {
      const int fb = 128 + 192 * k;
      #pragma unroll
      for (int nt = 0; nt < 4; ++nt) {
        v8f acc = {};
        #pragma unroll
        for (int kt = 0; kt < 6; ++kt) {
          v16bf a = loadA(fb, kt);
          v16bf b = loadB(lds_wv, WV_STRIDE, kt, nt);
          acc = __builtin_amdgcn_wmma_f32_16x16x32_bf16(false, a, false, b,
                                                        (short)0, acc, false, false);
        }
        const int col = 64 + 3 * (nt * 16 + row) + k;
        if (fullTile) {
          float* p = outBase + col;
          #pragma unroll
          for (int r = 0; r < 8; ++r) __builtin_nontemporal_store(acc[r], p + r * 256);
        } else {
          #pragma clang loop unroll(disable)
          for (int r = 0; r < 8; ++r) {
            const int zz = z0 + h * 8 + r;
            if (zz < n) out[(size_t)zz * 256 + col] = acc[r];
          }
        }
      }
    }
  }
}

extern "C" void kernel_launch(void* const* d_in, const int* in_sizes, int n_in,
                              void* d_out, int out_size, void* d_ws, size_t ws_size,
                              hipStream_t stream) {
  const float* x1     = (const float*)d_in[0];
  const float* x2     = (const float*)d_in[1];
  const float* w_ss_s = (const float*)d_in[2];
  const float* w_vv_s = (const float*)d_in[3];
  const float* w_sv_v = (const float*)d_in[4];
  const float* w_vs_v = (const float*)d_in[5];
  const float* w_vv_v = (const float*)d_in[6];
  float* out = (float*)d_out;

  const int n = in_sizes[0] / 256;
  const int numTiles = (n + 15) / 16;
  int blocks = (numTiles + WAVES_PER_BLOCK - 1) / WAVES_PER_BLOCK;
  if (blocks > 4096) blocks = 4096;
  if (blocks < 1) blocks = 1;

  tp_cdna5_kernel<<<blocks, 32 * WAVES_PER_BLOCK, 0, stream>>>(
      x1, x2, w_ss_s, w_vv_s, w_sv_v, w_vs_v, w_vv_v, out, n, numTiles);
}
